// GraphTRIADS_42992622633365
// MI455X (gfx1250) — compile-verified, matrix operand order checked by reading, not run on hardware
//
#include <hip/hip_runtime.h>
#include <hip/hip_bf16.h>

typedef __attribute__((ext_vector_type(16))) _Float16 v16h;
typedef __attribute__((ext_vector_type(8)))  _Float16 v8h;
typedef __attribute__((ext_vector_type(4)))  _Float16 v4h;
typedef __attribute__((ext_vector_type(8)))  float    v8f;

#define CAT16(a,b) __builtin_shufflevector(a,b,0,1,2,3,4,5,6,7,8,9,10,11,12,13,14,15)

static __device__ __forceinline__ float siluf(float x){ return x/(1.f+__expf(-x)); }
static __device__ __forceinline__ float sigmf(float x){ return 1.f/(1.f+__expf(-x)); }
static __device__ __forceinline__ float geluf(float x){ return 0.5f*x*(1.f+erff(x*0.7071067811865475f)); }

static __device__ __forceinline__ v4h cvt4(const float* src){
  float4 f = *(const float4*)src;
  v4h h; h[0]=(_Float16)f.x; h[1]=(_Float16)f.y; h[2]=(_Float16)f.z; h[3]=(_Float16)f.w;
  return h;
}

// ---------------------------------------------------------------- utilities
__global__ void fillz_k(float* p, int n){
  int i = blockIdx.x*blockDim.x + threadIdx.x;
  if (i < n) p[i] = 0.f;
}

// f32 [K,N] row-major -> f16 [N,Kpad] (transposed, zero-padded K)
__global__ void convert_wt_k(const float* W, _Float16* Wt, int K, int N, int Kpad){
  int idx = blockIdx.x*blockDim.x + threadIdx.x;
  if (idx >= N*Kpad) return;
  int n = idx / Kpad, k = idx - n*Kpad;
  Wt[idx] = (k < K) ? (_Float16)W[(size_t)k*N + n] : (_Float16)0.f;
}

// ---------------------------------------------------------------- embeddings
__global__ void embed_atoms_k(const int* z, const float* feat, const float* emb,
                              const float* afw, const float* afb, float* atoms){
  int idx = blockIdx.x*blockDim.x + threadIdx.x;
  if (idx >= 2048*256) return;
  int r = idx >> 8, d = idx & 255;
  int zz = z[r]; if (zz < 0) zz = 0; if (zz > 102) zz = 102;
  float s = emb[zz*256 + d] + afb[d];
  for (int k = 0; k < 18; ++k) s += feat[r*18 + k]*afw[k*256 + d];
  atoms[idx] = s;
}

__global__ void embed_bonds_k(const float* rbf, const float* vec, const float* phys,
                              const float* rw, const float* rb,
                              const float* vw, const float* vb,
                              const float* pw, const float* pb, float* bonds){
  int idx = blockIdx.x*blockDim.x + threadIdx.x;
  if (idx >= 65536*256) return;
  int e = idx >> 8, d = idx & 255;
  float a = rb[d]; for (int k = 0; k < 40; ++k) a += rbf[(size_t)e*40 + k]*rw[k*256 + d];
  float t = vb[d]; for (int k = 0; k < 3;  ++k) t += vec[(size_t)e*3  + k]*vw[k*256 + d];
  float p = pb[d]; for (int k = 0; k < 8;  ++k) p += phys[(size_t)e*8 + k]*pw[k*256 + d];
  bonds[idx] = a*tanhf(t) + p;
}

__global__ void mag_tok_k(const float* comp, const float* W, const float* bias,
                          const float* te0, float* comp_tok){
  int idx = blockIdx.x*blockDim.x + threadIdx.x;
  if (idx >= 32*22*256) return;
  int d = idx & 255; int bt = idx >> 8; int b = bt/22, t = bt - 22*b;
  float s = bias[d] + te0[d];
  for (int k = 0; k < 6; ++k) s += comp[b*400 + t*6 + k]*W[k*256 + d];
  comp_tok[(b*24 + t)*256 + d] = s;
}

__global__ void build_full_k(const float* comp_tok, const float* atoms, const float* te1, float* full){
  int idx = blockIdx.x*blockDim.x + threadIdx.x;
  if (idx >= 32*88*256) return;
  int d = idx & 255; int rs = idx >> 8; int b = rs/88, s = rs - 88*b;
  float v;
  if (s < 24) v = comp_tok[(b*24 + s)*256 + d];
  else        v = atoms[(b*64 + (s-24))*256 + d] + te1[d];
  full[idx] = v;
}

__global__ void split_full_k(const float* full, float* comp_tok, float* atoms){
  int idx = blockIdx.x*blockDim.x + threadIdx.x;
  if (idx >= 32*88*256) return;
  int d = idx & 255; int rs = idx >> 8; int b = rs/88, s = rs - 88*b;
  float v = full[idx];
  if (s < 24) comp_tok[(b*24 + s)*256 + d] = v;
  else        atoms[(b*64 + (s-24))*256 + d] = v;
}

__global__ void mean24_k(const float* comp_tok, float* xp){
  int idx = blockIdx.x*blockDim.x + threadIdx.x;
  if (idx >= 32*256) return;
  int b = idx >> 8, d = idx & 255;
  float s = 0.f;
  for (int t = 0; t < 24; ++t) s += comp_tok[(b*24 + t)*256 + d];
  xp[idx] = s*(1.f/24.f);
}

__global__ void gated_add_k(float* dst, const float* gate, const float* cand, int n){
  int i = blockIdx.x*blockDim.x + threadIdx.x;
  if (i < n) dst[i] += gate[i]*cand[i];
}

// generic tiny GEMM: out[r,c] = act( bias[c] + sum_k concat(s0,s1,s2)[r,k]*W[k,c] ) + addrow[c]
__global__ void small_mm_k(const float* s0, int w0, int ld0,
                           const float* s1, int w1, int ld1,
                           const float* s2, int w2, int ld2,
                           const float* W, const float* bias, const float* addrow,
                           float* out, int ldo, int M, int N, int act){
  int idx = blockIdx.x*blockDim.x + threadIdx.x;
  if (idx >= M*N) return;
  int r = idx / N, c = idx - r*N;
  float acc = bias ? bias[c] : 0.f;
  int ko = 0;
  for (int k = 0; k < w0; ++k) acc += s0[(size_t)r*ld0 + k]*W[(size_t)(ko + k)*N + c];
  ko += w0;
  if (s1){ for (int k = 0; k < w1; ++k) acc += s1[(size_t)r*ld1 + k]*W[(size_t)(ko + k)*N + c]; ko += w1; }
  if (s2){ for (int k = 0; k < w2; ++k) acc += s2[(size_t)r*ld2 + k]*W[(size_t)(ko + k)*N + c]; }
  if (act == 1) acc = siluf(acc);
  else if (act == 2) acc = sigmf(acc);
  out[(size_t)r*ldo + c] = acc + (addrow ? addrow[c] : 0.f);
}

// ---------------------------------------------------------------- WMMA GEMMs
// 16 waves (512 threads): one 16(M) x 256(N) tile per block, one shared A-tile
// gather/convert per K-step feeding 16 WMMA column tiles.  A-tile staged as
// float4 -> v4h chunks (segment boundaries are all multiples of 4).
template<int ACT> // 0 = none, 1 = exact GELU
__global__ __launch_bounds__(512)
void wmma_gemm_act(const float* s0, int w0, int ld0, const int* i0,
                   const float* s1, int w1, int ld1, const int* i1,
                   const float* s2, int w2, int ld2, const int* i2,
                   const _Float16* Wt, int Kpad,
                   const float* bias, float* C, int ldc, int M, int N){
  (void)M; (void)N;
  __shared__ __attribute__((aligned(16))) _Float16 As[16][40];
  __shared__ int r0s[16], r1s[16], r2s[16];
  const int tid = threadIdx.x, lane = tid & 31, wave = tid >> 5;
  const int row0 = blockIdx.x*16;
  const int n0 = blockIdx.y*256 + wave*16;
  if (tid < 16){
    int r = row0 + tid;
    r0s[tid] = i0 ? i0[r] : r;
    r1s[tid] = i1 ? i1[r] : r;
    r2s[tid] = i2 ? i2[r] : r;
  }
  v8f acc = {0.f,0.f,0.f,0.f,0.f,0.f,0.f,0.f};
  const int col = n0 + (lane & 15);
  const _Float16* wcol = Wt + (size_t)col*Kpad;
  const int klo = (lane < 16) ? 0 : 8;
  const int w01 = w0 + w1, Ktot = w01 + w2;
  for (int kb = 0; kb < Kpad; kb += 32){
    __syncthreads();
    if (tid < 128){                       // 16 rows x 8 chunks of 4
      int m = tid >> 3, c4 = (tid & 7)*4, k = kb + c4;
      v4h h = {(_Float16)0.f,(_Float16)0.f,(_Float16)0.f,(_Float16)0.f};
      const float* src = nullptr;
      if (k < w0)        src = s0 + (size_t)r0s[m]*ld0 + k;
      else if (k < w01)  src = s1 + (size_t)r1s[m]*ld1 + (k - w0);
      else if (k < Ktot) src = s2 + (size_t)r2s[m]*ld2 + (k - w01);
      if (src) h = cvt4(src);
      *(v4h*)&As[m][c4] = h;
    }
    if (kb + 128 < Kpad) __builtin_prefetch(wcol + kb + 128, 0, 1);
    __syncthreads();
    v8h a0 = *(const v8h*)&As[lane & 15][klo];
    v8h a1 = *(const v8h*)&As[lane & 15][klo + 16];
    v8h b0 = *(const v8h*)(wcol + kb + klo);
    v8h b1 = *(const v8h*)(wcol + kb + klo + 16);
    v16h a = CAT16(a0, a1), b = CAT16(b0, b1);
    acc = __builtin_amdgcn_wmma_f32_16x16x32_f16(false, a, false, b, (short)0, acc, false, false);
  }
  const int mb = (lane < 16) ? 0 : 8;
#pragma unroll
  for (int j = 0; j < 8; ++j){
    float v = acc[j] + (bias ? bias[col] : 0.f);
    if (ACT == 1) v = geluf(v);
    C[(size_t)(row0 + mb + j)*ldc + col] = v;
  }
}

// fused message kernel: msg = silu(A@Wm+bm) * sigmoid(A@Wg+bg), atomic scatter-add
// into agg[oidx[r]].  A row = [seg0[i0[r]](256) | seg1[i1[r]](256) | seg2[r](w2)]
// 16 waves: full N=256 covered per block -> gather A once.
__global__ __launch_bounds__(512)
void wmma_msg_scatter(const float* s0, int ld0, const int* i0,
                      const float* s1, int ld1, const int* i1,
                      const float* s2, int w2, int ld2,
                      const _Float16* WtM, const _Float16* WtG, int Kpad,
                      const float* bmb, const float* bgb,
                      const int* oidx, float* agg, int M){
  (void)M;
  __shared__ __attribute__((aligned(16))) _Float16 As[16][40];
  __shared__ int r0s[16], r1s[16], os[16];
  const int tid = threadIdx.x, lane = tid & 31, wave = tid >> 5;
  const int row0 = blockIdx.x*16;
  const int n0 = wave*16;
  if (tid < 16){
    int r = row0 + tid;
    r0s[tid] = i0[r]; r1s[tid] = i1[r]; os[tid] = oidx[r];
  }
  v8f accm = {0.f,0.f,0.f,0.f,0.f,0.f,0.f,0.f};
  v8f accg = {0.f,0.f,0.f,0.f,0.f,0.f,0.f,0.f};
  const int col = n0 + (lane & 15);
  const _Float16* wm = WtM + (size_t)col*Kpad;
  const _Float16* wg = WtG + (size_t)col*Kpad;
  const int klo = (lane < 16) ? 0 : 8;
  const int Ktot = 512 + w2;
  for (int kb = 0; kb < Kpad; kb += 32){
    __syncthreads();
    if (tid < 128){
      int m = tid >> 3, c4 = (tid & 7)*4, k = kb + c4;
      v4h h = {(_Float16)0.f,(_Float16)0.f,(_Float16)0.f,(_Float16)0.f};
      const float* src = nullptr;
      if (k < 256)       src = s0 + (size_t)r0s[m]*ld0 + k;
      else if (k < 512)  src = s1 + (size_t)r1s[m]*ld1 + (k - 256);
      else if (k < Ktot) src = s2 + (size_t)(row0 + m)*ld2 + (k - 512);
      if (src) h = cvt4(src);
      *(v4h*)&As[m][c4] = h;
    }
    if (kb + 128 < Kpad){
      __builtin_prefetch(wm + kb + 128, 0, 1);
      __builtin_prefetch(wg + kb + 128, 0, 1);
    }
    __syncthreads();
    v8h a0 = *(const v8h*)&As[lane & 15][klo];
    v8h a1 = *(const v8h*)&As[lane & 15][klo + 16];
    v16h a = CAT16(a0, a1);
    v8h m0 = *(const v8h*)(wm + kb + klo);
    v8h m1 = *(const v8h*)(wm + kb + klo + 16);
    v16h bm = CAT16(m0, m1);
    accm = __builtin_amdgcn_wmma_f32_16x16x32_f16(false, a, false, bm, (short)0, accm, false, false);
    v8h g0 = *(const v8h*)(wg + kb + klo);
    v8h g1 = *(const v8h*)(wg + kb + klo + 16);
    v16h bg = CAT16(g0, g1);
    accg = __builtin_amdgcn_wmma_f32_16x16x32_f16(false, a, false, bg, (short)0, accg, false, false);
  }
  const int mb = (lane < 16) ? 0 : 8;
#pragma unroll
  for (int j = 0; j < 8; ++j){
    float v = siluf(accm[j] + bmb[col]) * sigmf(accg[j] + bgb[col]);
    __hip_atomic_fetch_add(&agg[(size_t)os[mb + j]*256 + col], v,
                           __ATOMIC_RELAXED, __HIP_MEMORY_SCOPE_AGENT);
  }
}

// full-row (N=256, 16 waves) GEMM with fused LayerNorm epilogue.
// EPI 0: out = resid + silu(LN(x)*g+b)    (bond/atom update)
// EPI 1: out = LN(x + resid)*g + b        (post-attention / post-FF)
template<int EPI>
__global__ __launch_bounds__(512)
void wmma_gemm_ln(const float* s0, int w0, int ld0,
                  const float* s1, int w1, int ld1,
                  const _Float16* Wt, int Kpad,
                  const float* bias, const float* gam, const float* bet,
                  const float* resid, int ldr,
                  float* Out, int ldo, int M){
  (void)M;
  __shared__ __attribute__((aligned(16))) _Float16 As[16][40];
  __shared__ float xs[16][260];
  __shared__ float mu[16], rstd[16];
  const int tid = threadIdx.x, lane = tid & 31, wave = tid >> 5;
  const int row0 = blockIdx.x*16;
  const int n0 = wave*16;
  v8f acc = {0.f,0.f,0.f,0.f,0.f,0.f,0.f,0.f};
  const int col = n0 + (lane & 15);
  const _Float16* wcol = Wt + (size_t)col*Kpad;
  const int klo = (lane < 16) ? 0 : 8;
  const int w01 = w0 + w1;
  for (int kb = 0; kb < Kpad; kb += 32){
    __syncthreads();
    if (tid < 128){
      int m = tid >> 3, c4 = (tid & 7)*4, k = kb + c4;
      v4h h = {(_Float16)0.f,(_Float16)0.f,(_Float16)0.f,(_Float16)0.f};
      const float* src = nullptr;
      if (k < w0)       src = s0 + (size_t)(row0 + m)*ld0 + k;
      else if (k < w01) src = s1 + (size_t)(row0 + m)*ld1 + (k - w0);
      if (src) h = cvt4(src);
      *(v4h*)&As[m][c4] = h;
    }
    if (kb + 128 < Kpad) __builtin_prefetch(wcol + kb + 128, 0, 1);
    __syncthreads();
    v8h a0 = *(const v8h*)&As[lane & 15][klo];
    v8h a1 = *(const v8h*)&As[lane & 15][klo + 16];
    v8h b0 = *(const v8h*)(wcol + kb + klo);
    v8h b1 = *(const v8h*)(wcol + kb + klo + 16);
    v16h a = CAT16(a0, a1), b = CAT16(b0, b1);
    acc = __builtin_amdgcn_wmma_f32_16x16x32_f16(false, a, false, b, (short)0, acc, false, false);
  }
  const int mb = (lane < 16) ? 0 : 8;
#pragma unroll
  for (int j = 0; j < 8; ++j){
    float v = acc[j] + bias[col];
    if (EPI == 1) v += resid[(size_t)(row0 + mb + j)*ldr + col];
    xs[mb + j][col] = v;
  }
  __syncthreads();
  if (tid < 16){
    float s = 0.f, s2 = 0.f;
    for (int c = 0; c < 256; ++c){ float v = xs[tid][c]; s += v; s2 += v*v; }
    float m = s*(1.f/256.f);
    float var = s2*(1.f/256.f) - m*m;
    mu[tid] = m; rstd[tid] = rsqrtf(var + 1e-5f);
  }
  __syncthreads();
  for (int idx = tid; idx < 4096; idx += 512){
    int m = idx >> 8, c = idx & 255;
    float h = (xs[m][c] - mu[m])*rstd[m]*gam[c] + bet[c];
    float o;
    if (EPI == 0) o = resid[(size_t)(row0 + m)*ldr + c] + siluf(h);
    else          o = h;
    Out[(size_t)(row0 + m)*ldo + c] = o;
  }
}

// ---------------------------------------------------------------- attention
// one block per (batch, head); dh = 32; online softmax per query thread.
__global__ __launch_bounds__(128)
void attn_k(const float* qb, int qld, int qoff,
            const float* kvb, int kvld, int koff, int voff,
            float* ob, int oldc, int ooff,
            int Sq, int Sk, float scale){
  __shared__ float ks[88][33];
  __shared__ float vs[88][33];
  const int b = blockIdx.x, h = blockIdx.y, tid = threadIdx.x;
  for (int idx = tid; idx < Sk*32; idx += blockDim.x){
    int j = idx >> 5, d = idx & 31;
    ks[j][d] = kvb[(size_t)(b*Sk + j)*kvld + koff + h*32 + d];
    vs[j][d] = kvb[(size_t)(b*Sk + j)*kvld + voff + h*32 + d];
  }
  __syncthreads();
  if (tid < Sq){
    float qv[32], accv[32];
    const float* qp = qb + (size_t)(b*Sq + tid)*qld + qoff + h*32;
#pragma unroll
    for (int d = 0; d < 32; ++d){ qv[d] = qp[d]; accv[d] = 0.f; }
    float m = -3.0e38f, l = 0.f;
    for (int j = 0; j < Sk; ++j){
      float s = 0.f;
#pragma unroll
      for (int d = 0; d < 32; ++d) s += qv[d]*ks[j][d];
      s *= scale;
      float nm = fmaxf(m, s);
      float cf = __expf(m - nm), w = __expf(s - nm);
#pragma unroll
      for (int d = 0; d < 32; ++d) accv[d] = accv[d]*cf + w*vs[j][d];
      l = l*cf + w; m = nm;
    }
    float inv = 1.f/l;
    float* op = ob + (size_t)(b*Sq + tid)*oldc + ooff + h*32;
#pragma unroll
    for (int d = 0; d < 32; ++d) op[d] = accv[d]*inv;
  }
}

// ---------------------------------------------------------------- host side
static inline int cdiv(int a, int b){ return (a + b - 1)/b; }

enum {
  IN_COMP=0, IN_GPHYS, IN_ATOMZ, IN_ATOMF, IN_RBF, IN_VEC, IN_PHYS, IN_EIDX, IN_TRIP, IN_ANGLE,
  P_AFP_B=10, P_AFP_W, P_ATOM_EMBED,
  P_CA_IN_B, P_CA_IN_W, P_CA_OUT_B, P_CA_OUT_W,
  P_CA_N_B, P_CA_N_G, P_CTX_B, P_CTX_W, P_EXT_B, P_EXT_W,
  P_FF1_B, P_FF1_W, P_FF2_B, P_FF2_W,
  P_H1_B, P_H1_W, P_H2_B, P_H2_W,
  P_M2V_B, P_M2V_W, P_MAG_B, P_MAG_W,
  P_PHYS_B, P_PHYS_W, P_RBF_B, P_RBF_W,
  P_SA_IN_B, P_SA_IN_W, P_SA_OUT_B, P_SA_OUT_W,
  P_SA_FN_B, P_SA_FN_G, P_SA_N_B, P_SA_N_G,
  P_TRM_AG_B, P_TRM_AG_W, P_TRM_AM_B, P_TRM_AM_W,
  P_TRM_AU_B, P_TRM_AU_BE, P_TRM_AU_G, P_TRM_AU_W,
  P_TRM_BG_B, P_TRM_BG_W, P_TRM_BM_B, P_TRM_BM_W,
  P_TRM_BU_B, P_TRM_BU_BE, P_TRM_BU_G, P_TRM_BU_W,
  P_TYPE_EMBED, P_VEC_B, P_VEC_W,
  P_WU_AG_B, P_WU_AG_W, P_WU_AM_B, P_WU_AM_W,
  P_WU_AU_B, P_WU_AU_BE, P_WU_AU_G, P_WU_AU_W,
  P_WU_BG_B, P_WU_BG_W, P_WU_BM_B, P_WU_BM_W,
  P_WU_BU_B, P_WU_BU_BE, P_WU_BU_G, P_WU_BU_W,
  P_YG_B, P_YG_W, P_YU1_B, P_YU1_W, P_YU2_B, P_YU2_W,
  P_ZG_B, P_ZG_W, P_ZP_B, P_ZP_W, P_ZU1_B, P_ZU1_W, P_ZU2_B, P_ZU2_W
};

extern "C" void kernel_launch(void* const* d_in, const int* in_sizes, int n_in,
                              void* d_out, int out_size, void* d_ws, size_t ws_size,
                              hipStream_t stream){
  (void)in_sizes; (void)n_in; (void)out_size; (void)ws_size;
  const int Ec = 65536, Tt = 131072, Nn = 2048;
  const float ATTS = 0.17677669529663687f;
  #define FP(i) ((const float*)d_in[i])
  #define IPT(i) ((const int*)d_in[i])

  const float* comp  = FP(IN_COMP);
  const float* gphys = FP(IN_GPHYS);
  const int*   atomz = IPT(IN_ATOMZ);
  const float* atomf = FP(IN_ATOMF);
  const float* rbf   = FP(IN_RBF);
  const float* vecp  = FP(IN_VEC);
  const float* physp = FP(IN_PHYS);
  const int*   ei0   = IPT(IN_EIDX);
  const int*   ei1   = IPT(IN_EIDX) + Ec;
  const int*   tr0   = IPT(IN_TRIP);
  const int*   tr1   = IPT(IN_TRIP) + Tt;
  const float* angle = FP(IN_ANGLE);
  const float* te0   = FP(P_TYPE_EMBED);
  const float* te1   = te0 + 256;

  // workspace bump allocator
  char* base = (char*)d_ws; size_t off = 0;
  auto alloc = [&](size_t bytes)->char*{
    char* p = base + off; off = (off + bytes + 255) & ~(size_t)255; return p;
  };
  float* bonds    = (float*)alloc((size_t)Ec*256*4);
  float* agg_e    = (float*)alloc((size_t)Ec*256*4);
  float* atoms    = (float*)alloc((size_t)Nn*256*4);
  float* agg_n    = (float*)alloc((size_t)Nn*256*4);
  float* comp_tok = (float*)alloc((size_t)32*24*256*4);
  float* ctx      = (float*)alloc((size_t)32*256*4);
  float* full     = (float*)alloc((size_t)32*88*256*4);
  float* qkv      = (float*)alloc((size_t)32*88*768*4);
  float* attctx   = (float*)alloc((size_t)32*88*256*4);
  float* fftmp    = (float*)alloc((size_t)32*88*256*4);
  float* caq      = (float*)alloc((size_t)32*24*256*4);
  float* cakv     = (float*)alloc((size_t)32*64*512*4);
  float* cactx    = (float*)alloc((size_t)32*24*256*4);
  float* xp    = (float*)alloc(32*256*4);
  float* zbuf  = (float*)alloc(32*256*4);
  float* ybuf  = (float*)alloc(32*256*4);
  float* zinp  = (float*)alloc(32*256*4);
  float* tmp1  = (float*)alloc(32*256*4);
  float* cand  = (float*)alloc(32*256*4);
  float* gate  = (float*)alloc(32*256*4);
  float* ht    = (float*)alloc(32*128*4);
  _Float16* wu_bm_t  = (_Float16*)alloc(544*256*2);
  _Float16* wu_bg_t  = (_Float16*)alloc(544*256*2);
  _Float16* wu_bu_t  = (_Float16*)alloc(512*256*2);
  _Float16* wu_am_t  = (_Float16*)alloc(768*256*2);
  _Float16* wu_ag_t  = (_Float16*)alloc(768*256*2);
  _Float16* wu_au_t  = (_Float16*)alloc(512*256*2);
  _Float16* tm_bm_t  = (_Float16*)alloc(544*256*2);
  _Float16* tm_bg_t  = (_Float16*)alloc(544*256*2);
  _Float16* tm_bu_t  = (_Float16*)alloc(512*256*2);
  _Float16* tm_am_t  = (_Float16*)alloc(768*256*2);
  _Float16* tm_ag_t  = (_Float16*)alloc(768*256*2);
  _Float16* tm_au_t  = (_Float16*)alloc(512*256*2);
  _Float16* sa_in_t  = (_Float16*)alloc((size_t)768*256*2);
  _Float16* sa_out_t = (_Float16*)alloc(256*256*2);
  _Float16* ff1_t    = (_Float16*)alloc(256*256*2);
  _Float16* ff2_t    = (_Float16*)alloc(256*256*2);
  _Float16* ca_in_t  = (_Float16*)alloc((size_t)768*256*2);
  _Float16* ca_out_t = (_Float16*)alloc(256*256*2);

  auto conv = [&](int pidx, int K, int N, int Kpad, _Float16* dst){
    convert_wt_k<<<cdiv(N*Kpad,256),256,0,stream>>>(FP(pidx), dst, K, N, Kpad);
  };
  conv(P_WU_BM_W, 520, 256, 544, wu_bm_t);
  conv(P_WU_BG_W, 520, 256, 544, wu_bg_t);
  conv(P_WU_BU_W, 512, 256, 512, wu_bu_t);
  conv(P_WU_AM_W, 768, 256, 768, wu_am_t);
  conv(P_WU_AG_W, 768, 256, 768, wu_ag_t);
  conv(P_WU_AU_W, 512, 256, 512, wu_au_t);
  conv(P_TRM_BM_W, 520, 256, 544, tm_bm_t);
  conv(P_TRM_BG_W, 520, 256, 544, tm_bg_t);
  conv(P_TRM_BU_W, 512, 256, 512, tm_bu_t);
  conv(P_TRM_AM_W, 768, 256, 768, tm_am_t);
  conv(P_TRM_AG_W, 768, 256, 768, tm_ag_t);
  conv(P_TRM_AU_W, 512, 256, 512, tm_au_t);
  conv(P_SA_IN_W,  256, 768, 256, sa_in_t);
  conv(P_SA_OUT_W, 256, 256, 256, sa_out_t);
  conv(P_FF1_W,    256, 256, 256, ff1_t);
  conv(P_FF2_W,    256, 256, 256, ff2_t);
  conv(P_CA_IN_W,  256, 768, 256, ca_in_t);
  conv(P_CA_OUT_W, 256, 256, 256, ca_out_t);

  // embeddings
  embed_atoms_k<<<cdiv(Nn*256,256),256,0,stream>>>(atomz, atomf, FP(P_ATOM_EMBED),
                                                   FP(P_AFP_W), FP(P_AFP_B), atoms);
  embed_bonds_k<<<cdiv(Ec*256,256),256,0,stream>>>(rbf, vecp, physp,
                                                   FP(P_RBF_W), FP(P_RBF_B),
                                                   FP(P_VEC_W), FP(P_VEC_B),
                                                   FP(P_PHYS_W), FP(P_PHYS_B), bonds);

  // one message-passing layer
  auto run_mp = [&](const _Float16* bm_t, const _Float16* bg_t, const _Float16* bu_t,
                    const _Float16* am_t, const _Float16* ag_t, const _Float16* au_t,
                    int bm_b, int bg_b, int bu_b, int bu_g, int bu_be,
                    int am_b, int ag_b, int au_b, int au_g, int au_be){
    fillz_k<<<cdiv(Ec*256,256),256,0,stream>>>(agg_e, Ec*256);
    wmma_msg_scatter<<<dim3(Tt/16),512,0,stream>>>(
        bonds,256,tr0, bonds,256,tr1, angle,8,8,
        bm_t,bg_t,544, FP(bm_b),FP(bg_b), tr0, agg_e, Tt);
    wmma_gemm_ln<0><<<dim3(Ec/16),512,0,stream>>>(
        bonds,256,256, agg_e,256,256, bu_t,512,
        FP(bu_b),FP(bu_g),FP(bu_be), bonds,256, bonds,256, Ec);
    fillz_k<<<cdiv(Nn*256,256),256,0,stream>>>(agg_n, Nn*256);
    wmma_msg_scatter<<<dim3(Ec/16),512,0,stream>>>(
        atoms,256,ei0, atoms,256,ei1, bonds,256,256,
        am_t,ag_t,768, FP(am_b),FP(ag_b), ei1, agg_n, Ec);
    wmma_gemm_ln<0><<<dim3(Nn/16),512,0,stream>>>(
        atoms,256,256, agg_n,256,256, au_t,512,
        FP(au_b),FP(au_g),FP(au_be), atoms,256, atoms,256, Nn);
  };

  run_mp(wu_bm_t,wu_bg_t,wu_bu_t,wu_am_t,wu_ag_t,wu_au_t,
         P_WU_BM_B,P_WU_BG_B,P_WU_BU_B,P_WU_BU_G,P_WU_BU_BE,
         P_WU_AM_B,P_WU_AG_B,P_WU_AU_B,P_WU_AU_G,P_WU_AU_BE);

  // comp tokens + ctx
  mag_tok_k<<<cdiv(32*22*256,256),256,0,stream>>>(comp, FP(P_MAG_W), FP(P_MAG_B), te0, comp_tok);
  small_mm_k<<<cdiv(32*256,256),256,0,stream>>>(
      comp+132,68,400, nullptr,0,0, nullptr,0,0,
      FP(P_EXT_W), FP(P_EXT_B), te0, comp_tok + 22*256, 24*256, 32, 256, 0);
  small_mm_k<<<cdiv(32*256,256),256,0,stream>>>(
      comp+200,200,400, nullptr,0,0, nullptr,0,0,
      FP(P_M2V_W), FP(P_M2V_B), te0, comp_tok + 23*256, 24*256, 32, 256, 0);
  small_mm_k<<<cdiv(32*256,256),256,0,stream>>>(
      comp+200,11,400, gphys,16,16, nullptr,0,0,
      FP(P_CTX_W), FP(P_CTX_B), nullptr, ctx, 256, 32, 256, 0);
  fillz_k<<<cdiv(32*256,256),256,0,stream>>>(zbuf, 32*256);
  fillz_k<<<cdiv(32*256,256),256,0,stream>>>(ybuf, 32*256);

  for (int cyc = 0; cyc < 2; ++cyc){
    run_mp(tm_bm_t,tm_bg_t,tm_bu_t,tm_am_t,tm_ag_t,tm_au_t,
           P_TRM_BM_B,P_TRM_BG_B,P_TRM_BU_B,P_TRM_BU_G,P_TRM_BU_BE,
           P_TRM_AM_B,P_TRM_AG_B,P_TRM_AU_B,P_TRM_AU_G,P_TRM_AU_BE);

    build_full_k<<<cdiv(32*88*256,256),256,0,stream>>>(comp_tok, atoms, te1, full);

    // self-attention
    wmma_gemm_act<0><<<dim3(2816/16,3),512,0,stream>>>(
        full,256,256,nullptr, nullptr,0,0,nullptr, nullptr,0,0,nullptr,
        sa_in_t,256, FP(P_SA_IN_B), qkv,768, 2816,768);
    attn_k<<<dim3(32,8),128,0,stream>>>(qkv,768,0, qkv,768,256,512,
                                        attctx,256,0, 88,88, ATTS);
    wmma_gemm_ln<1><<<dim3(2816/16),512,0,stream>>>(
        attctx,256,256, nullptr,0,0, sa_out_t,256,
        FP(P_SA_OUT_B), FP(P_SA_N_G), FP(P_SA_N_B), full,256, full,256, 2816);
    // feed-forward
    wmma_gemm_act<1><<<dim3(2816/16,1),512,0,stream>>>(
        full,256,256,nullptr, nullptr,0,0,nullptr, nullptr,0,0,nullptr,
        ff1_t,256, FP(P_FF1_B), fftmp,256, 2816,256);
    wmma_gemm_ln<1><<<dim3(2816/16),512,0,stream>>>(
        fftmp,256,256, nullptr,0,0, ff2_t,256,
        FP(P_FF2_B), FP(P_SA_FN_G), FP(P_SA_FN_B), full,256, full,256, 2816);

    split_full_k<<<cdiv(32*88*256,256),256,0,stream>>>(full, comp_tok, atoms);

    // cross-attention (comp queries -> atom keys/values)
    wmma_gemm_act<0><<<dim3(768/16,1),512,0,stream>>>(
        comp_tok,256,256,nullptr, nullptr,0,0,nullptr, nullptr,0,0,nullptr,
        ca_in_t,256, FP(P_CA_IN_B), caq,256, 768,256);
    wmma_gemm_act<0><<<dim3(2048/16,2),512,0,stream>>>(
        atoms,256,256,nullptr, nullptr,0,0,nullptr, nullptr,0,0,nullptr,
        ca_in_t + (size_t)256*256, 256, FP(P_CA_IN_B)+256, cakv,512, 2048,512);
    attn_k<<<dim3(32,8),128,0,stream>>>(caq,256,0, cakv,512,0,256,
                                        cactx,256,0, 24,64, ATTS);
    wmma_gemm_ln<1><<<dim3(768/16),512,0,stream>>>(
        cactx,256,256, nullptr,0,0, ca_out_t,256,
        FP(P_CA_OUT_B), FP(P_CA_N_G), FP(P_CA_N_B), comp_tok,256, comp_tok,256, 768);

    mean24_k<<<cdiv(32*256,256),256,0,stream>>>(comp_tok, xp);

    // z update
    small_mm_k<<<cdiv(32*256,256),256,0,stream>>>(
        xp,256,256, ctx,256,256, ybuf,256,256,
        FP(P_ZP_W), FP(P_ZP_B), nullptr, zinp,256, 32,256, 0);
    small_mm_k<<<cdiv(32*256,256),256,0,stream>>>(
        zinp,256,256, zbuf,256,256, nullptr,0,0,
        FP(P_ZU1_W), FP(P_ZU1_B), nullptr, tmp1,256, 32,256, 1);
    small_mm_k<<<cdiv(32*256,256),256,0,stream>>>(
        tmp1,256,256, nullptr,0,0, nullptr,0,0,
        FP(P_ZU2_W), FP(P_ZU2_B), nullptr, cand,256, 32,256, 0);
    small_mm_k<<<cdiv(32*256,256),256,0,stream>>>(
        zinp,256,256, zbuf,256,256, nullptr,0,0,
        FP(P_ZG_W), FP(P_ZG_B), nullptr, gate,256, 32,256, 2);
    gated_add_k<<<cdiv(32*256,256),256,0,stream>>>(zbuf, gate, cand, 32*256);

    // y update
    small_mm_k<<<cdiv(32*256,256),256,0,stream>>>(
        ybuf,256,256, zbuf,256,256, nullptr,0,0,
        FP(P_YU1_W), FP(P_YU1_B), nullptr, tmp1,256, 32,256, 1);
    small_mm_k<<<cdiv(32*256,256),256,0,stream>>>(
        tmp1,256,256, nullptr,0,0, nullptr,0,0,
        FP(P_YU2_W), FP(P_YU2_B), nullptr, cand,256, 32,256, 0);
    small_mm_k<<<cdiv(32*256,256),256,0,stream>>>(
        ybuf,256,256, zbuf,256,256, nullptr,0,0,
        FP(P_YG_W), FP(P_YG_B), nullptr, gate,256, 32,256, 2);
    gated_add_k<<<cdiv(32*256,256),256,0,stream>>>(ybuf, gate, cand, 32*256);

    // head
    small_mm_k<<<cdiv(32*128,128),128,0,stream>>>(
        ybuf,256,256, nullptr,0,0, nullptr,0,0,
        FP(P_H1_W), FP(P_H1_B), nullptr, ht,128, 32,128, 1);
    small_mm_k<<<1,32,0,stream>>>(
        ht,128,128, nullptr,0,0, nullptr,0,0,
        FP(P_H2_W), FP(P_H2_B), nullptr, (float*)d_out,1, 32,1, 0);
  }
  #undef FP
  #undef IPT
}
